// NystromCausalBasis_42090679501208
// MI455X (gfx1250) — matrix operand order repaired and checked by reading, not compiled, not run on hardware
//
#include <hip/hip_runtime.h>

typedef _Float16 v16h __attribute__((ext_vector_type(16)));
typedef float    v8f  __attribute__((ext_vector_type(8)));
typedef float    f32x4 __attribute__((ext_vector_type(4)));

namespace {
constexpr int kB  = 16;
constexpr int kT  = 4096;
constexpr int kD  = 512;
constexpr int kR  = 256;
constexpr int kTM = 32;   // T rows per workgroup
constexpr int kNW = 8;    // waves per workgroup (256 threads, wave32)

constexpr size_t kSmemKs = (size_t)kR  * kD * sizeof(_Float16);  // 262144
constexpr size_t kSmemQs = (size_t)kTM * kD * sizeof(_Float16);  //  32768
constexpr size_t kSmemLM = (size_t)kR  * sizeof(int);            //   1024
constexpr size_t kSmemRS = (size_t)kNW * 16 * sizeof(float);     //    512
constexpr size_t kSmemTotal = kSmemKs + kSmemQs + kSmemLM + kSmemRS;
}

#define NCB_HAS_TDM __has_builtin(__builtin_amdgcn_tensor_load_to_lds)

// ---------------------------------------------------------------------------
// Kernel 1: gather landmark rows of z, L2-normalize, store as f16 K[b][r][d].
// ---------------------------------------------------------------------------
__global__ void nystrom_prep_k(const float* __restrict__ z,
                               const int*   __restrict__ lms,
                               _Float16*    __restrict__ Kf16) {
  const int br = blockIdx.x;       // b * R + r
  const int b  = br >> 8;          // R == 256
  const int r  = br & 255;
  const int t  = lms[r];
  const float* src = z + ((size_t)(b * kT + t)) * kD;

  const int tid = threadIdx.x;     // 256 threads; 2 elements each
  float x0 = src[tid];
  float x1 = src[tid + 256];

  __shared__ float red[256];
  red[tid] = x0 * x0 + x1 * x1;
  __syncthreads();
  for (int s = 128; s > 0; s >>= 1) {
    if (tid < s) red[tid] += red[tid + s];
    __syncthreads();
  }
  __shared__ float invn;
  if (tid == 0) invn = 1.0f / fmaxf(sqrtf(red[0]), 1e-12f);
  __syncthreads();
  const float inv = invn;

  _Float16* dst = Kf16 + (size_t)br * kD;
  dst[tid]       = (_Float16)(x0 * inv);
  dst[tid + 256] = (_Float16)(x1 * inv);
}

// ---------------------------------------------------------------------------
// Kernel 2: fused  normalize(Q tile) -> WMMA Q.K^T -> mask/poly/decay ->
//           row-normalize -> store [Phi | 1].
// grid = (T/kTM, B), block = 256 threads (8 waves).
// K (256 KB per batch) staged into LDS via the Tensor Data Mover.
// ---------------------------------------------------------------------------
__global__ void __launch_bounds__(256, 1)
nystrom_main(const float* __restrict__ z,
             const float* __restrict__ gw,
             const float* __restrict__ ta,
             const int*   __restrict__ lms,
             const _Float16* __restrict__ Kf16,
             float* __restrict__ out) {
  extern __shared__ char smem[];
  _Float16* Ks = (_Float16*)(smem);
  _Float16* Qs = (_Float16*)(smem + kSmemKs);
  int*      LM = (int*)     (smem + kSmemKs + kSmemQs);
  float*    RS = (float*)   (smem + kSmemKs + kSmemQs + kSmemLM); // [8][16]

  const int tid   = threadIdx.x;
  const int lane  = tid & 31;
  const int wid   = tid >> 5;
  const int b     = blockIdx.y;
  const int tBase = blockIdx.x * kTM;

  // ---- stage K (256 KB) into LDS ------------------------------------------
#if NCB_HAS_TDM
  typedef unsigned int u32x4 __attribute__((ext_vector_type(4)));
  typedef int          i32x4 __attribute__((ext_vector_type(4)));
  typedef int          i32x8 __attribute__((ext_vector_type(8)));
  if (wid == 0) {
    // Tensor DMA descriptor: 2D tensor == tile, 512 x 256 elements of 2 bytes.
    const unsigned long long ga =
        (unsigned long long)(const void*)(Kf16 + (size_t)b * kR * kD);
    const unsigned int ldsAddr = (unsigned int)(unsigned long long)(void*)Ks;
    // Group 0: count=1 | lds_addr | global_addr[56:0] | type=2 (bits 127:126)
    u32x4 g0 = { 1u,
                 ldsAddr,
                 (unsigned int)(ga & 0xFFFFFFFFull),
                 (unsigned int)(((ga >> 32) & 0x01FFFFFFull) | 0x80000000u) };
    // Group 1: data_size=1(2B); tensor_dim0=512; tensor_dim1=256;
    //          tile_dim0=512; tile_dim1=256; dim0_stride=512; dim1_stride=131072
    i32x8 g1 = { 0x00010000,          // workgroup_mask=0, data_size=1 (2 bytes)
                 0x02000000,          // tensor_dim0[15:0]=512 at bits [63:48]
                 0x01000000,          // tensor_dim0 hi=0, tensor_dim1[15:0]=256
                 0x02000000,          // tensor_dim1 hi=0, tile_dim0=512
                 0x00000100,          // tile_dim1=256, tile_dim2=0
                 0x00000200,          // tensor_dim0_stride[31:0]=512
                 0x00000000,          // stride0 hi=0, dim1_stride[15:0]=0
                 0x00000002 };        // dim1_stride[47:16]=2 -> 131072
    i32x4 gz = {0, 0, 0, 0};
#if __clang_major__ >= 23
    i32x8 gz8 = {0, 0, 0, 0, 0, 0, 0, 0};
    __builtin_amdgcn_tensor_load_to_lds(g0, g1, gz, gz, gz8, 0);
#else
    __builtin_amdgcn_tensor_load_to_lds(g0, g1, gz, gz, 0);
#endif
  }
#else
  {
    const uint4* ksrc = (const uint4*)(Kf16 + (size_t)b * kR * kD);
    uint4* kdst = (uint4*)Ks;
    constexpr int nIter = (int)(kSmemKs / 16 / 256);  // 64
#pragma unroll 4
    for (int j = 0; j < nIter; ++j)
      kdst[tid + j * 256] = ksrc[tid + j * 256];
  }
#endif
  if (tid < kR) LM[tid] = lms[tid];

  // ---- stage + normalize 32 Q rows into LDS as f16 (overlaps the DMA) -----
  // wave w handles rows w*4 .. w*4+3; lane owns 16 contiguous elements.
  for (int rr = 0; rr < 4; ++rr) {
    const int row = wid * 4 + rr;
    const int t   = tBase + row;
    const f32x4* s4 = (const f32x4*)(z + ((size_t)(b * kT + t)) * kD + lane * 16);
    f32x4 q0 = __builtin_nontemporal_load(s4 + 0);
    f32x4 q1 = __builtin_nontemporal_load(s4 + 1);
    f32x4 q2 = __builtin_nontemporal_load(s4 + 2);
    f32x4 q3 = __builtin_nontemporal_load(s4 + 3);
    float v[16] = {q0[0], q0[1], q0[2], q0[3], q1[0], q1[1], q1[2], q1[3],
                   q2[0], q2[1], q2[2], q2[3], q3[0], q3[1], q3[2], q3[3]};
    float ss = 0.0f;
#pragma unroll
    for (int i = 0; i < 16; ++i) ss = fmaf(v[i], v[i], ss);
    // wave32 all-reduce
#pragma unroll
    for (int m = 16; m >= 1; m >>= 1) ss += __shfl_xor(ss, m, 32);
    const float inv = 1.0f / fmaxf(sqrtf(ss), 1e-12f);
    union { _Float16 h[16]; uint4 u[2]; } pk;
#pragma unroll
    for (int i = 0; i < 16; ++i) pk.h[i] = (_Float16)(v[i] * inv);
    uint4* qd = (uint4*)(Qs + row * kD + lane * 16);
    qd[0] = pk.u[0];
    qd[1] = pk.u[1];
  }

#if NCB_HAS_TDM
  if (wid == 0) __builtin_amdgcn_s_wait_tensorcnt(0);  // K tile landed in LDS
#endif
  __syncthreads();

  // ---- per-thread scalar prep (softmax of gw, softplus of ta) -------------
  const float g0s = gw[0], g1s = gw[1], g2s = gw[2];
  const float gm = fmaxf(g0s, fmaxf(g1s, g2s));
  const float e0 = __expf(g0s - gm), e1 = __expf(g1s - gm), e2 = __expf(g2s - gm);
  const float esum = e0 + e1 + e2;
  const float a0 = e0 / esum, a1 = e1 / esum, a2 = e2 / esum;
  const float tav  = ta[0];
  const float adec = fmaxf(tav, 0.0f) + log1pf(__expf(-fabsf(tav)));
  const float invT = 1.0f / (float)kT;

  // ---- WMMA main loop -----------------------------------------------------
  // wave -> (row tile rt in {0,1}, column-tile group cg in {0..3} of 4 tiles)
  const int rt  = wid & 1;
  const int cg  = wid >> 1;
  const int hi  = lane >> 4;   // half-wave select
  const int m16 = lane & 15;

  union F16x16 { v16h h; uint4 u[2]; };

  v8f acc[4];
  const v8f vzero = {0.f, 0.f, 0.f, 0.f, 0.f, 0.f, 0.f, 0.f};
#pragma unroll
  for (int jj = 0; jj < 4; ++jj) acc[jj] = vzero;

  const _Float16* aBase = Qs + (rt * 16 + m16) * kD;

  for (int kc = 0; kc < kD / 32; ++kc) {   // 16 K-chunks of 32
    // A fragment (16x32 f16): half-wave hi -> K base 0 or 8; pairs at +0,+16
    F16x16 af;
    const _Float16* ap = aBase + kc * 32 + hi * 8;
    af.u[0] = *(const uint4*)(ap);
    af.u[1] = *(const uint4*)(ap + 16);
#pragma unroll
    for (int jj = 0; jj < 4; ++jj) {
      const int n = (cg * 4 + jj) * 16 + m16;
      // B fragment (32x16 f16): lane holds column n, half-wave hi -> K base 0 or 8; pairs at +0,+16
      const _Float16* bp = Ks + n * kD + kc * 32 + hi * 8;
      F16x16 bf;
      bf.u[0] = *(const uint4*)(bp);
      bf.u[1] = *(const uint4*)(bp + 16);
      acc[jj] = __builtin_amdgcn_wmma_f32_16x16x32_f16(
          false, af.h, false, bf.h, (short)0, acc[jj], false, false);
    }
  }

  // ---- fused epilogue on C fragments --------------------------------------
  // C/D layout: lane holds N = lane&15; element v -> row M = v + 8*(lane>=16)
  float rs[8];
#pragma unroll
  for (int v = 0; v < 8; ++v) rs[v] = 0.0f;

#pragma unroll
  for (int jj = 0; jj < 4; ++jj) {
    const int r  = (cg * 4 + jj) * 16 + m16;
    const int lm = LM[r];
#pragma unroll
    for (int v = 0; v < 8; ++v) {
      const int t  = tBase + rt * 16 + v + hi * 8;
      const float c  = (lm < t) ? acc[jj][v] : 0.0f;  // causal mask on C
      const float c2 = c * c;
      const float P1 = 0.5f * (1.0f + c);
      const float P2 = 0.5f * (3.0f * c2 - 1.0f);
      const float P3 = 0.5f * (c * (5.0f * c2 - 3.0f));
      float phi = a0 * P1 + a1 * P2 + a2 * P3;
      const float dist = fmaxf((float)(t - lm), 0.0f) * invT;
      phi *= __expf(-adec * dist);
      acc[jj][v] = phi;
      rs[v] += phi;
    }
  }

  // row-sum: butterfly across the 16-lane group (same rows), then LDS partials
#pragma unroll
  for (int v = 0; v < 8; ++v) {
    float s = rs[v];
    s += __shfl_xor(s, 1, 32);
    s += __shfl_xor(s, 2, 32);
    s += __shfl_xor(s, 4, 32);
    s += __shfl_xor(s, 8, 32);
    rs[v] = s;
  }
  if (m16 == 0) {
#pragma unroll
    for (int v = 0; v < 8; ++v) RS[wid * 16 + v + hi * 8] = rs[v];
  }
  __syncthreads();

  // ---- normalize and store (non-temporal: write-once stream) --------------
#pragma unroll
  for (int v = 0; v < 8; ++v) {
    const int rowInTile = rt * 16 + v + hi * 8;
    const int rloc = v + hi * 8;
    const float total = RS[(rt + 0) * 16 + rloc] + RS[(rt + 2) * 16 + rloc] +
                        RS[(rt + 4) * 16 + rloc] + RS[(rt + 6) * 16 + rloc];
    const float invSum = 1.0f / fmaxf(total, 1e-6f);
    const int t = tBase + rowInTile;
    float* orow = out + ((size_t)(b * kT + t)) * (kR + 1);
#pragma unroll
    for (int jj = 0; jj < 4; ++jj) {
      const int r = (cg * 4 + jj) * 16 + m16;
      __builtin_nontemporal_store(acc[jj][v] * invSum, orow + r);
    }
  }
  // ones column
  if (tid < kTM) {
    const int t = tBase + tid;
    __builtin_nontemporal_store(1.0f,
        out + ((size_t)(b * kT + t)) * (kR + 1) + kR);
  }
}

// ---------------------------------------------------------------------------
extern "C" void kernel_launch(void* const* d_in, const int* in_sizes, int n_in,
                              void* d_out, int out_size, void* d_ws, size_t ws_size,
                              hipStream_t stream) {
  (void)in_sizes; (void)n_in; (void)out_size; (void)ws_size;
  const float* z   = (const float*)d_in[0];
  const float* gw  = (const float*)d_in[1];
  const float* ta  = (const float*)d_in[2];
  const int*   lms = (const int*)d_in[3];
  float* out = (float*)d_out;
  _Float16* Kf16 = (_Float16*)d_ws;   // 16*256*512 f16 = 4 MB

  nystrom_prep_k<<<dim3(kB * kR), dim3(256), 0, stream>>>(z, lms, Kf16);

  dim3 grid(kT / kTM, kB);
  nystrom_main<<<grid, dim3(256), (int)kSmemTotal, stream>>>(
      z, gw, ta, lms, Kf16, out);
}
